// ResBlock_17910013624550
// MI455X (gfx1250) — 8-bit matrix operand order repaired and checked by reading, compiled, not run on hardware
//
#include <hip/hip_runtime.h>
#include <cstdint>

typedef int v8i __attribute__((ext_vector_type(8)));

#define CC 256
#define BN 8
#define HH 128
#define WW 128

__global__ __launch_bounds__(256)
void k_binarize_nchw_to_nhwc(const float* __restrict__ x, int8_t* __restrict__ xb)
{
    __shared__ int8_t tile[32][33];
    int bid = blockIdx.x;
    int wt = bid & 3;  bid >>= 2;
    int ct = bid & 7;  bid >>= 3;
    int n  = bid >> 7;
    int h  = bid & 127;
    int lane = threadIdx.x & 31;
    int row  = threadIdx.x >> 5;
#pragma unroll
    for (int i = 0; i < 4; ++i) {
        int c = ct*32 + row*4 + i;
        float v = x[(((size_t)n*CC + c)*HH + h)*WW + wt*32 + lane];
        tile[row*4 + i][lane] = (v >= 0.0f) ? (int8_t)1 : (int8_t)-1;
    }
    __syncthreads();
#pragma unroll
    for (int i = 0; i < 4; ++i) {
        int w = wt*32 + row*4 + i;
        xb[(((size_t)n*HH + h)*WW + w)*CC + ct*32 + lane] = tile[lane][row*4 + i];
    }
}

__device__ __forceinline__ int pack4(int b0, int b1, int b2, int b3) {
    return (b0 & 0xff) | ((b1 & 0xff) << 8) | ((b2 & 0xff) << 16) | ((b3 & 0xff) << 24);
}

__global__ __launch_bounds__(256)
void k_pack_branch_w(const float* __restrict__ w1, const float* __restrict__ w2,
                     const float* __restrict__ w3, const float* __restrict__ w4,
                     int8_t* __restrict__ bfrag)
{
    int id = blockIdx.x*256 + threadIdx.x;
    if (id >= 4*5*4*4*32*8) return;
    int dw = id & 7;    id >>= 3;
    int lane = id & 31; id >>= 5;
    int nt = id & 3;    id >>= 2;
    int kb = id & 3;    id >>= 2;
    int t  = id % 5;
    int br = id / 5;
    const float* w = (br==0) ? w1 : (br==1) ? w2 : (br==2) ? w3 : w4;
    int o = nt*16 + (lane & 15);
    int kbase = (dw >> 1)*16 + ((lane >= 16) ? 8 : 0) + (dw & 1)*4;
    int b[4];
#pragma unroll
    for (int q = 0; q < 4; ++q) {
        int c = kb*64 + kbase + q;
        float v = w[((size_t)o*CC + c)*5 + t];
        b[q] = (v >= 0.0f) ? 1 : -1;
    }
    int frag = ((br*5 + t)*4 + kb)*4 + nt;
    *(int*)(bfrag + (size_t)frag*1024 + lane*32 + dw*4) = pack4(b[0], b[1], b[2], b[3]);
}

__global__ __launch_bounds__(256)
void k_pack_wc(const float* __restrict__ wc, int8_t* __restrict__ wfrag)
{
    int id = blockIdx.x*256 + threadIdx.x;
    if (id >= 36*16*32*8) return;
    int dw = id & 7;    id >>= 3;
    int lane = id & 31; id >>= 5;
    int nt = id & 15;   id >>= 4;
    int ks = id;
    int kb = ks & 3;
    int tap = ks >> 2;
    int kh = tap / 3, kw = tap % 3;
    int o = nt*16 + (lane & 15);
    int kbase = (dw >> 1)*16 + ((lane >= 16) ? 8 : 0) + (dw & 1)*4;
    int b[4];
#pragma unroll
    for (int q = 0; q < 4; ++q) {
        int c = kb*64 + kbase + q;
        float v = wc[((size_t)o*CC + c)*9 + kh*3 + kw];
        b[q] = (v >= 0.0f) ? 1 : -1;
    }
    *(int*)(wfrag + ((size_t)ks*16 + nt)*1024 + lane*32 + dw*4) = pack4(b[0], b[1], b[2], b[3]);
}

__global__ __launch_bounds__(128)
void k_branch_conv(const int8_t* __restrict__ xb, const int8_t* __restrict__ bfrag,
                   const float* __restrict__ b1, const float* __restrict__ b2,
                   const float* __restrict__ b3, const float* __restrict__ b4,
                   const float* __restrict__ a1, int8_t* __restrict__ yb)
{
    __shared__ __align__(16) int8_t strip[136*CC];
    const int br = blockIdx.y;
    const int d  = (br & 1) ? 2 : 1;
    const bool htaps = (br >= 2);
    int line = blockIdx.x;
    int n  = line >> 7;
    int hw = line & 127;
    const int tidb = threadIdx.x;
    const int lane = tidb & 31;
    const int wv   = tidb >> 5;
    const int plen = 128 + 4*d;

    const uint32_t ldsbase = (uint32_t)(uintptr_t)&strip[0];
    const int sub = tidb & 15;
    for (int p = tidb >> 4; p < plen; p += 8) {
        int gp = p - 2*d;
        if (gp >= 0 && gp < 128) {
            size_t gpix = htaps ? (((size_t)n*HH + gp)*WW + hw)
                                : (((size_t)n*HH + hw)*WW + gp);
            uint32_t la = ldsbase + (uint32_t)(p*CC + sub*16);
            uint64_t ga = (uint64_t)(uintptr_t)(xb + gpix*CC + sub*16);
            asm volatile("global_load_async_to_lds_b128 %0, %1, off"
                         :: "v"(la), "v"(ga) : "memory");
        } else {
            *(int4*)&strip[p*CC + sub*16] = make_int4(0,0,0,0);
        }
    }
    asm volatile("s_wait_asynccnt 0x0" ::: "memory");
    __syncthreads();

    v8i a00 = {}, a01 = {}, a02 = {}, a03 = {};
    v8i a10 = {}, a11 = {}, a12 = {}, a13 = {};
    const int M = lane & 15;
    const int koff = (lane >= 16) ? 8 : 0;
    const int kba[8] = {0,4,16,20,32,36,48,52};
    const int8_t* fb = bfrag + (size_t)br*81920 + lane*32;

    v8i B0 = *(const v8i*)(fb);
    v8i B1 = *(const v8i*)(fb + 1024);
    v8i B2 = *(const v8i*)(fb + 2048);
    v8i B3 = *(const v8i*)(fb + 3072);
    for (int s = 0; s < 20; ++s) {
        v8i nB0 = B0, nB1 = B1, nB2 = B2, nB3 = B3;
        if (s < 19) {
            const int8_t* f = fb + (size_t)(s + 1)*4096;
            nB0 = *(const v8i*)(f);
            nB1 = *(const v8i*)(f + 1024);
            nB2 = *(const v8i*)(f + 2048);
            nB3 = *(const v8i*)(f + 3072);
        }
        int t = s >> 2, kb = s & 3;
        int pb = wv*32 + M + t*d;
        v8i A;
#pragma unroll
        for (int j = 0; j < 8; ++j)
            A[j] = *(const int*)&strip[pb*CC + kb*64 + kba[j] + koff];
        a00 = __builtin_amdgcn_wmma_i32_16x16x64_iu8(true, A, true, B0, a00, false, false);
        a01 = __builtin_amdgcn_wmma_i32_16x16x64_iu8(true, A, true, B1, a01, false, false);
        a02 = __builtin_amdgcn_wmma_i32_16x16x64_iu8(true, A, true, B2, a02, false, false);
        a03 = __builtin_amdgcn_wmma_i32_16x16x64_iu8(true, A, true, B3, a03, false, false);
#pragma unroll
        for (int j = 0; j < 8; ++j)
            A[j] = *(const int*)&strip[(pb + 16)*CC + kb*64 + kba[j] + koff];
        a10 = __builtin_amdgcn_wmma_i32_16x16x64_iu8(true, A, true, B0, a10, false, false);
        a11 = __builtin_amdgcn_wmma_i32_16x16x64_iu8(true, A, true, B1, a11, false, false);
        a12 = __builtin_amdgcn_wmma_i32_16x16x64_iu8(true, A, true, B2, a12, false, false);
        a13 = __builtin_amdgcn_wmma_i32_16x16x64_iu8(true, A, true, B3, a13, false, false);
        B0 = nB0; B1 = nB1; B2 = nB2; B3 = nB3;
    }

    const float* bias = (br==0) ? b1 : (br==1) ? b2 : (br==2) ? b3 : b4;
    const float alpha = a1[0];
    const int nn = lane & 15;
    const int mo = (lane >= 16) ? 8 : 0;
    v8i accs[2][4] = {{a00, a01, a02, a03}, {a10, a11, a12, a13}};
#pragma unroll
    for (int mt = 0; mt < 2; ++mt) {
        const int m0 = wv*32 + mt*16;
#pragma unroll
        for (int nt = 0; nt < 4; ++nt) {
            int o = nt*16 + nn;
            float bo = bias[o];
#pragma unroll
            for (int r = 0; r < 8; ++r) {
                int m = r + mo;
                float z = (float)accs[mt][nt][r] + bo;
                float pl = (z >= 0.0f) ? z : alpha*z;
                int pix = m0 + m;
                size_t gpix = htaps ? (((size_t)n*HH + pix)*WW + hw)
                                    : (((size_t)n*HH + hw)*WW + pix);
                yb[gpix*CC + br*64 + o] = (pl >= 0.0f) ? (int8_t)1 : (int8_t)-1;
            }
        }
    }
}

__global__ __launch_bounds__(64)
void k_conv3x3(const int8_t* __restrict__ yb, const int8_t* __restrict__ wfrag,
               const float* __restrict__ bc, const float* __restrict__ a2,
               const float* __restrict__ x, float* __restrict__ out)
{
    __shared__ __align__(16) int8_t strip[3*66*CC];
    const int g = blockIdx.y;
    int tid  = blockIdx.x;
    int t64  = tid & 1;
    int line = tid >> 1;
    int n = line >> 7, h = line & 127;
    const int w0base = t64*64;
    const int tidb = threadIdx.x;
    const int lane = tidb & 31;
    const int wv   = tidb >> 5;

    const uint32_t ldsbase = (uint32_t)(uintptr_t)&strip[0];
    const int sub = tidb & 15;
    for (int idx = tidb >> 4; idx < 3*66; idx += 4) {
        int row = idx / 66, col = idx - row*66;
        int hh = h + row - 1, ww = w0base - 1 + col;
        if (hh >= 0 && hh < HH && ww >= 0 && ww < WW) {
            size_t gpix = ((size_t)n*HH + hh)*WW + ww;
            uint32_t la = ldsbase + (uint32_t)(idx*CC + sub*16);
            uint64_t ga = (uint64_t)(uintptr_t)(yb + gpix*CC + sub*16);
            asm volatile("global_load_async_to_lds_b128 %0, %1, off"
                         :: "v"(la), "v"(ga) : "memory");
        } else {
            *(int4*)&strip[idx*CC + sub*16] = make_int4(0,0,0,0);
        }
    }
    asm volatile("s_wait_asynccnt 0x0" ::: "memory");
    __syncthreads();

    v8i a00 = {}, a01 = {}, a02 = {}, a03 = {};
    v8i a10 = {}, a11 = {}, a12 = {}, a13 = {};
    const int M = lane & 15;
    const int koff = (lane >= 16) ? 8 : 0;
    const int kba[8] = {0,4,16,20,32,36,48,52};
    const int8_t* fb = wfrag + (size_t)g*4096 + lane*32;

    v8i B0 = *(const v8i*)(fb);
    v8i B1 = *(const v8i*)(fb + 1024);
    v8i B2 = *(const v8i*)(fb + 2048);
    v8i B3 = *(const v8i*)(fb + 3072);
    for (int s = 0; s < 36; ++s) {
        v8i nB0 = B0, nB1 = B1, nB2 = B2, nB3 = B3;
        if (s < 35) {
            const int8_t* f = fb + (size_t)(s + 1)*16384;
            nB0 = *(const v8i*)(f);
            nB1 = *(const v8i*)(f + 1024);
            nB2 = *(const v8i*)(f + 2048);
            nB3 = *(const v8i*)(f + 3072);
        }
        int tap = s >> 2, kb = s & 3;
        int kh = tap / 3, kw = tap - kh*3;
        int col = wv*32 + M + kw;
        v8i A;
#pragma unroll
        for (int j = 0; j < 8; ++j)
            A[j] = *(const int*)&strip[(kh*66 + col)*CC + kb*64 + kba[j] + koff];
        a00 = __builtin_amdgcn_wmma_i32_16x16x64_iu8(true, A, true, B0, a00, false, false);
        a01 = __builtin_amdgcn_wmma_i32_16x16x64_iu8(true, A, true, B1, a01, false, false);
        a02 = __builtin_amdgcn_wmma_i32_16x16x64_iu8(true, A, true, B2, a02, false, false);
        a03 = __builtin_amdgcn_wmma_i32_16x16x64_iu8(true, A, true, B3, a03, false, false);
#pragma unroll
        for (int j = 0; j < 8; ++j)
            A[j] = *(const int*)&strip[(kh*66 + col + 16)*CC + kb*64 + kba[j] + koff];
        a10 = __builtin_amdgcn_wmma_i32_16x16x64_iu8(true, A, true, B0, a10, false, false);
        a11 = __builtin_amdgcn_wmma_i32_16x16x64_iu8(true, A, true, B1, a11, false, false);
        a12 = __builtin_amdgcn_wmma_i32_16x16x64_iu8(true, A, true, B2, a12, false, false);
        a13 = __builtin_amdgcn_wmma_i32_16x16x64_iu8(true, A, true, B3, a13, false, false);
        B0 = nB0; B1 = nB1; B2 = nB2; B3 = nB3;
    }

    const float alpha = a2[0];
    const int nn = lane & 15;
    const int mo = (lane >= 16) ? 8 : 0;
    v8i accs[2][4] = {{a00, a01, a02, a03}, {a10, a11, a12, a13}};
#pragma unroll
    for (int mt = 0; mt < 2; ++mt) {
        const int w0 = w0base + wv*32 + mt*16;
#pragma unroll
        for (int nt = 0; nt < 4; ++nt) {
            int o = g*64 + nt*16 + nn;
            float bo = bc[o];
            size_t obase = (((size_t)n*CC + o)*HH + h)*WW + w0;
#pragma unroll
            for (int r = 0; r < 8; ++r) {
                int m = r + mo;
                float z = (float)accs[mt][nt][r] + bo;
                float pl = (z >= 0.0f) ? z : alpha*z;
                out[obase + m] = pl + x[obase + m];
            }
        }
    }
}

extern "C" void kernel_launch(void* const* d_in, const int* in_sizes, int n_in,
                              void* d_out, int out_size, void* d_ws, size_t ws_size,
                              hipStream_t stream)
{
    (void)in_sizes; (void)n_in; (void)out_size; (void)ws_size;
    const float* x  = (const float*)d_in[0];
    const float* w1 = (const float*)d_in[1];
    const float* b1 = (const float*)d_in[2];
    const float* w2 = (const float*)d_in[3];
    const float* b2 = (const float*)d_in[4];
    const float* w3 = (const float*)d_in[5];
    const float* b3 = (const float*)d_in[6];
    const float* w4 = (const float*)d_in[7];
    const float* b4 = (const float*)d_in[8];
    const float* a1 = (const float*)d_in[9];
    const float* wc = (const float*)d_in[10];
    const float* bc = (const float*)d_in[11];
    const float* a2 = (const float*)d_in[12];

    int8_t* xb    = (int8_t*)d_ws;
    int8_t* ybuf  = xb + (size_t)33554432;
    int8_t* bfrag = ybuf + (size_t)33554432;
    int8_t* wfrag = bfrag + (size_t)327680;

    k_binarize_nchw_to_nhwc<<<32768, 256, 0, stream>>>(x, xb);
    k_pack_branch_w<<<(4*5*4*4*32*8 + 255)/256, 256, 0, stream>>>(w1, w2, w3, w4, bfrag);
    k_pack_wc<<<(36*16*32*8 + 255)/256, 256, 0, stream>>>(wc, wfrag);

    dim3 gb(1024, 4);
    k_branch_conv<<<gb, 128, 0, stream>>>(xb, bfrag, b1, b2, b3, b4, a1, ybuf);

    dim3 gc(2048, 4);
    k_conv3x3<<<gc, 64, 0, stream>>>(ybuf, wfrag, bc, a2, x, (float*)d_out);
}
